// GCNEncoder_4123168604875
// MI455X (gfx1250) — matrix operand order repaired and checked by reading, not compiled, not run on hardware
//
#include <hip/hip_runtime.h>
#include <hip/hip_bf16.h>
#include <stdint.h>

typedef __attribute__((ext_vector_type(16))) __bf16 v16bf;
typedef __attribute__((ext_vector_type(8)))  float  v8f;
typedef __attribute__((ext_vector_type(4)))  unsigned int u32x4;
typedef __attribute__((ext_vector_type(8)))  unsigned int u32x8;

#define DEV static __device__ __forceinline__

DEV unsigned short f2bf(float f) {
    unsigned int u = __float_as_uint(f);
    unsigned int r = u + 0x7FFFu + ((u >> 16) & 1u);   // round-to-nearest-even
    return (unsigned short)(r >> 16);
}

DEV void storev(float* p, float v)          { *p = v; }
DEV void storev(unsigned short* p, float v) { *p = f2bf(v); }

// ---------------------------------------------------------------- utilities
__global__ void fill_f32(float* p, float v, int n) {
    int i = blockIdx.x * blockDim.x + threadIdx.x;
    if (i < n) p[i] = v;
}

__global__ void deg_count(const int* __restrict__ dst, float* __restrict__ deg, int E) {
    int e = blockIdx.x * blockDim.x + threadIdx.x;
    if (e < E) atomicAdd(&deg[dst[e]], 1.0f);
}

__global__ void make_dinv(const float* __restrict__ deg, float* __restrict__ dinv, int n) {
    int i = blockIdx.x * blockDim.x + threadIdx.x;
    if (i < n) {
        float d = deg[i];                       // >= 1 (self loop included)
        dinv[i] = d > 0.0f ? rsqrtf(d) : 0.0f;
    }
}

// W [K,F] fp32 row-major  ->  Wt [F,K] bf16 (column-major of W): contiguous-K loads
__global__ void w_to_bf16_t(const float* __restrict__ W, unsigned short* __restrict__ Wt,
                            int K, int F) {
    int i = blockIdx.x * blockDim.x + threadIdx.x;
    if (i < K * F) {
        int k = i / F, c = i % F;
        Wt[c * K + k] = f2bf(W[i]);
    }
}

// ---------------------------------------------------------- BN statistics
__global__ void col_stats(const float* __restrict__ x, float* __restrict__ sum,
                          float* __restrict__ sumsq, int n, int F) {
    __shared__ float ss[256];
    __shared__ float sq[256];
    int tid = threadIdx.x;
    int c = tid % F;
    int rpb = 256 / F;                 // rows per block iteration (2 or 4)
    int r = blockIdx.x * rpb + tid / F;
    int stride = gridDim.x * rpb;
    float s = 0.0f, s2 = 0.0f;
    for (; r < n; r += stride) {
        float v = x[(size_t)r * F + c];
        s += v; s2 += v * v;
    }
    ss[tid] = s; sq[tid] = s2;
    __syncthreads();
    if (tid < F) {
        for (int j = 1; j < rpb; ++j) { s += ss[tid + j * F]; s2 += sq[tid + j * F]; }
        atomicAdd(&sum[c], s);
        atomicAdd(&sumsq[c], s2);
    }
}

// y = (x - mean) * gamma * rsqrt(var + eps) + beta  [optional relu], OUT = f32 or bf16
template <typename OUT, bool RELU>
__global__ void bn_apply(const float* __restrict__ x, const float* __restrict__ sum,
                         const float* __restrict__ sumsq, const float* __restrict__ gamma,
                         const float* __restrict__ beta, OUT* __restrict__ out,
                         long long total, int F, float invN) {
    long long i = (long long)blockIdx.x * blockDim.x + threadIdx.x;
    if (i >= total) return;
    int c = (int)(i % F);
    float m  = sum[c] * invN;
    float vr = fmaxf(sumsq[c] * invN - m * m, 0.0f);   // biased variance
    float sc = gamma[c] * rsqrtf(vr + 1e-5f);
    float v  = (x[i] - m) * sc + beta[c];
    if (RELU) v = fmaxf(v, 0.0f);
    storev(&out[i], v);
}

// ---------------------------------------------------------------- GEMM (WMMA + TDM)
// C[n,F] = A[n,128] * B[128,F] (+bias, relu). A bf16 row-major, Bt bf16 [F,128].
// One block = one 16-row stripe. Wave 0 DMAs the 16x128 bf16 A tile into LDS with
// the Tensor Data Mover (tensor_load_to_lds, TENSORcnt), then each wave32 computes
// one 16x16 tile with 4x v_wmma_f32_16x16x32_bf16, reading A fragments from LDS.
template <bool RELU, bool BF16OUT>
__global__ void gemm_wmma(const unsigned short* __restrict__ A,
                          const unsigned short* __restrict__ Bt,
                          const float* __restrict__ bias,
                          float* __restrict__ outF, unsigned short* __restrict__ outB,
                          int n, int F, int tilesPerRow) {
    __shared__ unsigned short atile[16 * 128];          // 4KB bf16 tile
    const int K = 128;
    const int lane = threadIdx.x & 31;
    const int wave = threadIdx.x >> 5;
    const int r0 = blockIdx.x * 16;
    if (r0 >= n) return;                                 // block-uniform

    if (wave == 0) {
        // --- TDM descriptor (ISA 8.3/8.4): 2D tile, 2-byte elements ---
        unsigned long long ga = (unsigned long long)(uintptr_t)(A + (size_t)r0 * K);
        unsigned ldsoff = (unsigned)(uintptr_t)(&atile[0]);   // low 32 bits = LDS offset
        u32x4 g0 = { 1u,                                  // count=1 (valid), no gather
                     ldsoff,                              // lds_addr
                     (unsigned)ga,                        // global_addr[31:0]
                     ((unsigned)(ga >> 32) & 0x01FFFFFFu) // global_addr[56:32]
                       | 0x80000000u };                   // type=2 ("image")
        u32x8 g1 = { (1u << 16),                          // data_size=1 -> 2 bytes
                     (128u << 16),                        // tensor_dim0[15:0] = 128
                     (16u << 16),                         // tensor_dim1[15:0] = 16
                     (128u << 16),                        // tile_dim0 = 128
                     16u,                                 // tile_dim1 = 16
                     128u,                                // tensor_dim0_stride = 128
                     0u, 0u };
        asm volatile("tensor_load_to_lds %0, %1" :: "s"(g0), "s"(g1) : "memory");
        __builtin_amdgcn_s_wait_tensorcnt(0);
    }
    __syncthreads();

    const int ctile = wave % tilesPerRow;
    const int c0 = ctile * 16;
    const int col = c0 + (lane & 15);
    const int kA = (lane >> 4) << 3;                     // 0 | 8
    const int kB = (lane >> 4) << 3;                     // 0 | 8

    const unsigned short* arow = atile + (size_t)(lane & 15) * K;   // LDS
    const unsigned short* bcol = Bt + (size_t)col * K;              // global (L2-resident)

    v8f acc = {};
#pragma unroll
    for (int kb = 0; kb < 128; kb += 32) {
        union { v16bf v; uint4 q[2]; } ua, ub;
        const uint4* pa = reinterpret_cast<const uint4*>(arow + kb + kA);
        ua.q[0] = pa[0];                                 // K = kb+kA    .. +7   (ds_load_b128)
        ua.q[1] = pa[2];                                 // K = kb+kA+16 .. +23
        const uint4* pb = reinterpret_cast<const uint4*>(bcol + kb + kB);
        ub.q[0] = pb[0];                                 // K = kb+kB    .. +7
        ub.q[1] = pb[2];                                 // K = kb+kB+16 .. +23
        acc = __builtin_amdgcn_wmma_f32_16x16x32_bf16(false, ua.v, false, ub.v,
                                                      (short)0, acc, false, false);
    }

    const int rb = r0 + ((lane >> 4) << 3);
    const float bv = bias ? bias[col] : 0.0f;
#pragma unroll
    for (int v = 0; v < 8; ++v) {
        float val = acc[v] + bv;
        if (RELU) val = fmaxf(val, 0.0f);
        size_t o = (size_t)(rb + v) * F + col;
        if (BF16OUT) outB[o] = f2bf(val); else outF[o] = val;
    }
}

// ------------------------------------------------------------- edge scatter
// out[dst] += xw[src] * dinv[src]*dinv[dst].  One thread = 4 features of one edge;
// F=128 -> one wave per edge (fully coalesced 512B gather).
__global__ void scatter_edges(const float* __restrict__ xw, const int* __restrict__ src,
                              const int* __restrict__ dst, const float* __restrict__ dinv,
                              float* __restrict__ out, int E, int F, int fqShift) {
    long long idx = (long long)blockIdx.x * blockDim.x + threadIdx.x;
    const int fq = 1 << fqShift;                        // F/4
    if (idx >= (long long)E * fq) return;
    int e = (int)(idx >> fqShift);
    int q = (int)(idx & (fq - 1));
    int s = src[e], d = dst[e];
    float w = dinv[s] * dinv[d];
    const float4 v = *reinterpret_cast<const float4*>(xw + (size_t)s * F + q * 4);
    float* o = out + (size_t)d * F + q * 4;
    atomicAdd(o + 0, v.x * w);
    atomicAdd(o + 1, v.y * w);
    atomicAdd(o + 2, v.z * w);
    atomicAdd(o + 3, v.w * w);
}

// self-loop term + bias:  out[i,c] += xw[i,c]*dinv[i]^2 + b[c]
__global__ void finalize_conv(float* __restrict__ out, const float* __restrict__ xw,
                              const float* __restrict__ dinv, const float* __restrict__ bias,
                              long long total, int F, int fShift) {
    long long i = (long long)blockIdx.x * blockDim.x + threadIdx.x;
    if (i >= total) return;
    int node = (int)(i >> fShift);
    int c = (int)(i & (F - 1));
    float w = dinv[node];
    out[i] += xw[i] * w * w + bias[c];
}

// ================================================================= launcher
extern "C" void kernel_launch(void* const* d_in, const int* in_sizes, int n_in,
                              void* d_out, int out_size, void* d_ws, size_t ws_size,
                              hipStream_t stream) {
    const int N = 50000, E = 640000;
    const int F_H = 128, F_OUT = 64;

    const float* x        = (const float*)d_in[0];
    const int*   ei       = (const int*)d_in[1];
    const int*   src      = ei;
    const int*   dst      = ei + E;
    const float* bn_in_g  = (const float*)d_in[2];
    const float* bn_in_b  = (const float*)d_in[3];
    const float* bn1_g    = (const float*)d_in[4];
    const float* bn1_b    = (const float*)d_in[5];
    const float* bn2_g    = (const float*)d_in[6];
    const float* bn2_b    = (const float*)d_in[7];
    const float* proj_W   = (const float*)d_in[8];
    const float* proj_b   = (const float*)d_in[9];
    const float* conv1_W  = (const float*)d_in[10];
    const float* conv1_b  = (const float*)d_in[11];
    const float* conv2_W  = (const float*)d_in[12];
    const float* conv2_b  = (const float*)d_in[13];
    float* out = (float*)d_out;

    // workspace carve-out (256B aligned slices)
    char* p = (char*)d_ws;
    auto carve = [&](size_t bytes) -> char* {
        char* r = p;
        p += (bytes + 255) & ~(size_t)255;
        return r;
    };
    float*          deg   = (float*)carve((size_t)N * 4);
    float*          dinv  = (float*)carve((size_t)N * 4);
    float*          sum   = (float*)carve(128 * 4);          // contiguous with sumsq
    float*          sumsq = (float*)carve(128 * 4);
    unsigned short* WtP   = (unsigned short*)carve(128 * 128 * 2);
    unsigned short* Wt1   = (unsigned short*)carve(128 * 128 * 2);
    unsigned short* Wt2   = (unsigned short*)carve(128 * 64 * 2);
    unsigned short* hbn   = (unsigned short*)carve((size_t)N * 128 * 2); // BN(x); reused as h2
    unsigned short* h1    = (unsigned short*)carve((size_t)N * 128 * 2);
    float*          xw    = (float*)carve((size_t)N * 128 * 4);          // GEMM out (reused)
    float*          agg   = (float*)carve((size_t)N * 128 * 4);          // scatter accum (reused)

    const int T = 256;
    auto nb = [](long long n, int t) { return (unsigned)((n + t - 1) / t); };

    // ---- degrees (with self loops) + dinv
    fill_f32<<<nb(N, T), T, 0, stream>>>(deg, 1.0f, N);
    deg_count<<<nb(E, T), T, 0, stream>>>(dst, deg, E);
    make_dinv<<<nb(N, T), T, 0, stream>>>(deg, dinv, N);

    // ---- weight transpose + bf16 conversion
    w_to_bf16_t<<<nb(128 * 128, T), T, 0, stream>>>(proj_W, WtP, 128, 128);
    w_to_bf16_t<<<nb(128 * 128, T), T, 0, stream>>>(conv1_W, Wt1, 128, 128);
    w_to_bf16_t<<<nb(128 * 64, T), T, 0, stream>>>(conv2_W, Wt2, 128, 64);

    const float invN = 1.0f / (float)N;

    // ---- stage 0: BN(x) -> bf16
    fill_f32<<<1, T, 0, stream>>>(sum, 0.0f, 256);           // zeros sum+sumsq
    col_stats<<<512, T, 0, stream>>>(x, sum, sumsq, N, 128);
    bn_apply<unsigned short, false><<<nb((long long)N * 128, T), T, 0, stream>>>(
        x, sum, sumsq, bn_in_g, bn_in_b, hbn, (long long)N * 128, 128, invN);

    // ---- stage 1: h1 = relu(hbn @ proj_W + b)   [bf16 out]
    gemm_wmma<true, true><<<3125, 256, 0, stream>>>(hbn, WtP, proj_b, nullptr, h1,
                                                    N, F_H, 8);

    // ---- conv1: xw = h1 @ conv1_W ; aggregate ; +selfloop+bias ; BN+relu -> h2(bf16)
    gemm_wmma<false, false><<<3125, 256, 0, stream>>>(h1, Wt1, nullptr, xw, nullptr,
                                                      N, F_H, 8);
    fill_f32<<<nb((long long)N * 128, T), T, 0, stream>>>(agg, 0.0f, N * 128);
    scatter_edges<<<nb((long long)E * 32, T), T, 0, stream>>>(xw, src, dst, dinv, agg,
                                                              E, F_H, 5);
    finalize_conv<<<nb((long long)N * 128, T), T, 0, stream>>>(agg, xw, dinv, conv1_b,
                                                               (long long)N * 128, F_H, 7);
    fill_f32<<<1, T, 0, stream>>>(sum, 0.0f, 256);
    col_stats<<<512, T, 0, stream>>>(agg, sum, sumsq, N, 128);
    bn_apply<unsigned short, true><<<nb((long long)N * 128, T), T, 0, stream>>>(
        agg, sum, sumsq, bn1_g, bn1_b, hbn /* h2 */, (long long)N * 128, 128, invN);

    // ---- conv2: xw2 = h2 @ conv2_W ; aggregate ; +selfloop+bias ; BN -> d_out
    gemm_wmma<false, false><<<3125, 128, 0, stream>>>(hbn, Wt2, nullptr, xw, nullptr,
                                                      N, F_OUT, 4);
    fill_f32<<<nb((long long)N * 64, T), T, 0, stream>>>(agg, 0.0f, N * 64);
    scatter_edges<<<nb((long long)E * 16, T), T, 0, stream>>>(xw, src, dst, dinv, agg,
                                                              E, F_OUT, 4);
    finalize_conv<<<nb((long long)N * 64, T), T, 0, stream>>>(agg, xw, dinv, conv2_b,
                                                              (long long)N * 64, F_OUT, 6);
    fill_f32<<<1, T, 0, stream>>>(sum, 0.0f, 256);
    col_stats<<<512, T, 0, stream>>>(agg, sum, sumsq, N, 64);
    bn_apply<float, false><<<nb((long long)N * 64, T), T, 0, stream>>>(
        agg, sum, sumsq, bn2_g, bn2_b, out, (long long)N * 64, 64, invN);
}
